// LSHAttention_33998961115891
// MI455X (gfx1250) — compile-verified, matrix operand order checked by reading, not run on hardware
//
#include <hip/hip_runtime.h>
#include <hip/hip_bf16.h>

// ---------------------------------------------------------------------------
// Self-attention module for MI455X (gfx1250):
//   qk = x@Wq^T + bq (Q==K), v = x@Wv^T + bv
//   out = softmax_causal(QK^T/8) @ V, then @Wo^T + bo
// bf16 WMMA (f32 accumulate) everywhere; TDM (tensor_load_to_lds) staging of
// X tiles with double buffering; flash-softmax keeps scores out of HBM.
// ---------------------------------------------------------------------------

typedef __bf16 bf16_t;
typedef bf16_t v16bf __attribute__((ext_vector_type(16)));
typedef bf16_t v8bf  __attribute__((ext_vector_type(8)));
typedef float  v8f   __attribute__((ext_vector_type(8)));

typedef unsigned int tdm_u32x4 __attribute__((ext_vector_type(4)));
typedef int          tdm_i32x8 __attribute__((ext_vector_type(8)));
typedef int          tdm_i32x4 __attribute__((ext_vector_type(4)));

#define SEQ    2048
#define HID    1024
#define NHEAD  16
#define DK     64
#define NBATCH 2
#define MTOT   (NBATCH * SEQ)   // 4096 tokens

static __device__ __forceinline__ bf16_t f2bf(float f) {
    unsigned u = __builtin_bit_cast(unsigned, f);
    unsigned r = (u + 0x7FFFu + ((u >> 16) & 1u)) >> 16;   // RNE
    unsigned short s = (unsigned short)r;
    return __builtin_bit_cast(bf16_t, s);
}

static __device__ __forceinline__ v8f wmma_bf16(v16bf a, v16bf b, v8f c) {
    return __builtin_amdgcn_wmma_f32_16x16x32_bf16(
        false, a, false, b, (short)0, c, false, false);
}

// 16x32 A-fragment (or mirrored 32x16 B-fragment, lane = N column) from a
// row-major bf16 matrix (global or LDS). Per CDNA5 ISA:
//   lanes 0-15 : row = lane,    K = {k0+0..7,  k0+16..23}
//   lanes 16-31: row = lane-16, K = {k0+8..15, k0+24..31}
static __device__ __forceinline__ v16bf load_frag(const bf16_t* base, int ld,
                                                  int row0, int k0, int lane) {
    const int hi = lane >> 4;
    const int r  = row0 + (lane & 15);
    const bf16_t* p = base + r * ld + k0 + hi * 8;
    union { v16bf v; v8bf h[2]; } u;
    u.h[0] = *(const v8bf*)(p);
    u.h[1] = *(const v8bf*)(p + 16);
    return u.v;
}

// ---------------------------------------------------------------------------
// TDM: issue a 2-D tile load Global -> LDS via Tensor DMA Descriptor (D#).
// D# bit packing per CDNA5 ISA:
//   g0: count=1 | lds_addr[63:32] | global_addr[120:64] | type=2[127:126]
//   g1: data_size[17:16]=1 (2B) | tensor_dim0[79:48] | tensor_dim1[111:80]
//       | tile_dim0[127:112] | tile_dim1[143:128] | tensor_dim0_stride[207:160]
// Tracked by TENSORcnt (wave-level op; EXEC ignored).
// This toolchain exposes the 6-arg builtin (g0, g1, g2, g3, g4, cpol).
// ---------------------------------------------------------------------------
static __device__ __forceinline__ void tdm_load_tile_2d(
    unsigned lds_byte_addr, const bf16_t* gaddr,
    unsigned tensor_w, unsigned tensor_h,
    unsigned tile_w, unsigned tile_h, unsigned row_stride) {
    unsigned long long ga = (unsigned long long)(size_t)gaddr;
    tdm_u32x4 g0;
    g0[0] = 1u;                                            // count=1 (valid user D#)
    g0[1] = lds_byte_addr;                                 // lds_addr
    g0[2] = (unsigned)(ga & 0xFFFFFFFFu);                  // global_addr[31:0]
    g0[3] = (unsigned)((ga >> 32) & 0x01FFFFFFu)           // global_addr[56:32]
          | 0x80000000u;                                   // type=2 ("image")
    tdm_i32x8 g1;
    g1[0] = (int)(1u << 16);                               // data_size=1 -> 2 bytes
    g1[1] = (int)(tensor_w << 16);                         // tensor_dim0 lo16
    g1[2] = (int)((tensor_w >> 16) | (tensor_h << 16));    // dim0 hi16 | dim1 lo16
    g1[3] = (int)((tensor_h >> 16) | (tile_w << 16));      // dim1 hi16 | tile_dim0
    g1[4] = (int)(tile_h & 0xFFFFu);                       // tile_dim1
    g1[5] = (int)row_stride;                               // tensor_dim0_stride lo32
    g1[6] = 0;
    g1[7] = 0;
    tdm_i32x4 g2 = (tdm_i32x4){0, 0, 0, 0};
    tdm_i32x4 g3 = (tdm_i32x4){0, 0, 0, 0};
    tdm_i32x8 g4 = (tdm_i32x8){0, 0, 0, 0, 0, 0, 0, 0};
    __builtin_amdgcn_tensor_load_to_lds(g0, g1, g2, g3, g4, 0);
}

// ---------------------------------------------------------------------------
// Kernel 0: f32 -> bf16 conversion, vectorized (float4 in, ushort4 out).
// ---------------------------------------------------------------------------
__global__ void f32_to_bf16_kernel(const float* __restrict__ src,
                                   bf16_t* __restrict__ dst, int n4) {
    int i = blockIdx.x * blockDim.x + threadIdx.x;
    int stride = gridDim.x * blockDim.x;
    for (; i < n4; i += stride) {
        float4 f = ((const float4*)src)[i];
        ushort4 o;
        o.x = __builtin_bit_cast(unsigned short, f2bf(f.x));
        o.y = __builtin_bit_cast(unsigned short, f2bf(f.y));
        o.z = __builtin_bit_cast(unsigned short, f2bf(f.z));
        o.w = __builtin_bit_cast(unsigned short, f2bf(f.w));
        ((ushort4*)dst)[i] = o;
    }
}

// ---------------------------------------------------------------------------
// Kernel 1: projection GEMM  Y = X @ W^T + bias, 32x64 tile per wave.
// X tiles (32x32 bf16, 2 KiB) staged into LDS by the Tensor Data Mover with
// double buffering; W fragments direct from global (L2-resident, 2 MiB).
// Last K-iteration is peeled so s_wait_tensorcnt gets constant immediates.
//   mode 0: write Q layout [b, h, s, d]   (bf16)
//   mode 1: write V^T layout [b, h, d, s] (bf16)
// ---------------------------------------------------------------------------
__global__ __launch_bounds__(128) void proj_gemm_kernel(
    const bf16_t* __restrict__ X, const bf16_t* __restrict__ W,
    const float* __restrict__ bias, bf16_t* __restrict__ Y, int mode) {
    __shared__ __align__(16) bf16_t Xlds[4][2][32 * 32];   // 16 KiB / block

    const int lane = threadIdx.x & 31;
    const int widx = threadIdx.x >> 5;
    const int wg   = blockIdx.x * 4 + widx;          // 2048 waves total
    const int nt   = wg >> 7;                        // 0..15  (64-col tile)
    const int mt   = wg & 127;                       // 0..127 (32-row tile)
    const int m0   = mt * 32;
    const int n0   = nt * 64;

    v8f acc[2][4];
#pragma unroll
    for (int r = 0; r < 2; ++r)
#pragma unroll
        for (int t = 0; t < 4; ++t) acc[r][t] = (v8f){0,0,0,0,0,0,0,0};

    auto do_step = [&](int k0, int cur, bool pf) {
        const v16bf a0 = load_frag(&Xlds[widx][cur][0], 32, 0,  0, lane);
        const v16bf a1 = load_frag(&Xlds[widx][cur][0], 32, 16, 0, lane);
#pragma unroll
        for (int t = 0; t < 4; ++t) {
            if (pf)   // prefetch next-k chunk of this lane's W row
                __builtin_prefetch(W + (n0 + t * 16 + (lane & 15)) * HID + k0 + 32, 0, 3);
            const v16bf b = load_frag(W, HID, n0 + t * 16, k0, lane);
            acc[0][t] = wmma_bf16(a0, b, acc[0][t]);
            acc[1][t] = wmma_bf16(a1, b, acc[1][t]);
        }
    };

    // Pre-issue first X tile via TDM.
    tdm_load_tile_2d((unsigned)(size_t)&Xlds[widx][0][0], X + m0 * HID,
                     HID, MTOT, 32, 32, HID);

    // Steady state: issue next tile, wait for current (<=1 outstanding), compute.
    for (int k0 = 0; k0 < HID - 32; k0 += 32) {
        const int cur = (k0 >> 5) & 1;
        tdm_load_tile_2d((unsigned)(size_t)&Xlds[widx][cur ^ 1][0],
                         X + m0 * HID + (k0 + 32), HID, MTOT, 32, 32, HID);
        __builtin_amdgcn_s_wait_tensorcnt(1);
        __asm__ volatile("" ::: "memory");
        do_step(k0, cur, true);
    }
    // Peeled last iteration: wait for everything.
    __builtin_amdgcn_s_wait_tensorcnt(0);
    __asm__ volatile("" ::: "memory");
    do_step(HID - 32, ((HID - 32) >> 5) & 1, false);

    // Write-out. C layout: element v of lane -> (m = v + 8*(lane>=16), n = lane&15).
    const int hi = lane >> 4;
    const int nl = lane & 15;
#pragma unroll
    for (int t = 0; t < 4; ++t) {
        const int n  = n0 + t * 16 + nl;
        const float bv = bias[n];
        const int h = n >> 6, d = n & 63;
#pragma unroll
        for (int r = 0; r < 2; ++r) {
#pragma unroll
            for (int v = 0; v < 8; ++v) {
                const int m = m0 + r * 16 + v + hi * 8;
                const int b = m >> 11, s = m & (SEQ - 1);
                const float val = acc[r][t][v] + bv;
                int idx;
                if (mode == 0) idx = (((b * NHEAD + h) * SEQ + s) * DK + d);
                else           idx = (((b * NHEAD + h) * DK + d) * SEQ + s);
                Y[idx] = f2bf(val);
            }
        }
    }
}

// ---------------------------------------------------------------------------
// Kernel 2: fused causal flash attention. One wave per (b, h, 16-query tile).
//   Qg : [b,h,s,d] bf16 (used as both Q rows and K rows)
//   Vtg: [b,h,d,s] bf16 (transposed V for contiguous P@V B-fragments)
//   AO : [b, s, h*64+d] bf16 (token-major for the output GEMM)
// Keys in blocks of 32 (one full WMMA K-dim for P@V). Scores never touch HBM.
// ---------------------------------------------------------------------------
__global__ __launch_bounds__(128) void flash_attn_kernel(
    const bf16_t* __restrict__ Qg, const bf16_t* __restrict__ Vtg,
    bf16_t* __restrict__ AO) {
    __shared__ __align__(16) bf16_t Plds[4][16][32];   // per-wave P staging

    const int lane = threadIdx.x & 31;
    const int widx = threadIdx.x >> 5;
    const int wg   = blockIdx.x * 4 + widx;            // 4096 waves
    const int qt   = wg & 127;
    const int h    = (wg >> 7) & (NHEAD - 1);
    const int b    = wg >> 11;

    const bf16_t* Qb = Qg  + (b * NHEAD + h) * SEQ * DK;
    const bf16_t* Vb = Vtg + (b * NHEAD + h) * DK * SEQ;
    const int q0 = qt * 16;
    const int hi = lane >> 4;
    const int nl = lane & 15;

    const v16bf QA0 = load_frag(Qb, DK, q0, 0, lane);
    const v16bf QA1 = load_frag(Qb, DK, q0, 32, lane);

    v8f O[4];
#pragma unroll
    for (int t = 0; t < 4; ++t) O[t] = (v8f){0,0,0,0,0,0,0,0};
    float mrun[8], lrun[8];
#pragma unroll
    for (int v = 0; v < 8; ++v) { mrun[v] = -__builtin_inff(); lrun[v] = 0.0f; }

    const int nkb = (q0 >> 5) + 1;                     // causal 32-key blocks
    for (int kb = 0; kb < nkb; ++kb) {
        const int kk0 = kb * 32;
        if (kb + 1 < nkb) {                            // prefetch next key block
            __builtin_prefetch(Qb + (kk0 + 32 + lane) * DK, 0, 3);
            __builtin_prefetch(Vb + (lane & 15) * SEQ + kk0 + 32, 0, 3);
        }

        // S[16x32] = Q(16x64) @ K^T(64x32): two 16x16 C tiles, chained over d.
        v8f S0 = (v8f){0,0,0,0,0,0,0,0};
        v8f S1 = (v8f){0,0,0,0,0,0,0,0};
        S0 = wmma_bf16(QA0, load_frag(Qb, DK, kk0,      0,  lane), S0);
        S0 = wmma_bf16(QA1, load_frag(Qb, DK, kk0,      32, lane), S0);
        S1 = wmma_bf16(QA0, load_frag(Qb, DK, kk0 + 16, 0,  lane), S1);
        S1 = wmma_bf16(QA1, load_frag(Qb, DK, kk0 + 16, 32, lane), S1);

        // Scale, causal mask, online softmax (row = v + 8*hi, key col = nl).
        float p0[8], p1[8], csc[8];
#pragma unroll
        for (int v = 0; v < 8; ++v) {
            const int q = q0 + v + hi * 8;
            float s0 = S0[v] * 0.125f;                 // 1/sqrt(64)
            float s1 = S1[v] * 0.125f;
            if (kk0 + nl      > q) s0 = -__builtin_inff();
            if (kk0 + 16 + nl > q) s1 = -__builtin_inff();

            float rm = fmaxf(s0, s1);
            rm = fmaxf(rm, __shfl_xor(rm, 1, 32));
            rm = fmaxf(rm, __shfl_xor(rm, 2, 32));
            rm = fmaxf(rm, __shfl_xor(rm, 4, 32));
            rm = fmaxf(rm, __shfl_xor(rm, 8, 32));

            const float mnew = fmaxf(mrun[v], rm);
            const float cs   = __expf(mrun[v] - mnew);
            const float e0   = __expf(s0 - mnew);
            const float e1   = __expf(s1 - mnew);
            float rs = e0 + e1;
            rs += __shfl_xor(rs, 1, 32);
            rs += __shfl_xor(rs, 2, 32);
            rs += __shfl_xor(rs, 4, 32);
            rs += __shfl_xor(rs, 8, 32);

            lrun[v] = lrun[v] * cs + rs;
            mrun[v] = mnew;
            p0[v] = e0; p1[v] = e1; csc[v] = cs;
        }

#pragma unroll
        for (int t = 0; t < 4; ++t)
#pragma unroll
            for (int v = 0; v < 8; ++v) O[t][v] *= csc[v];

        // Stage P (C layout -> LDS row-major 16x32), reload as A-fragment.
#pragma unroll
        for (int v = 0; v < 8; ++v) {
            Plds[widx][v + hi * 8][nl]      = f2bf(p0[v]);
            Plds[widx][v + hi * 8][nl + 16] = f2bf(p1[v]);
        }
        __asm__ volatile("" ::: "memory");   // LDS is in-order per wave
        const v16bf PA = load_frag(&Plds[widx][0][0], 32, 0, 0, lane);
        __asm__ volatile("" ::: "memory");

        // O[16x64] += P(16x32) @ V(32x64); B lane = d column, contiguous in V^T.
#pragma unroll
        for (int t = 0; t < 4; ++t)
            O[t] = wmma_bf16(PA, load_frag(Vb, SEQ, t * 16, kk0, lane), O[t]);
    }

    // Normalize and write [b, s, h*64 + d] as bf16.
#pragma unroll
    for (int t = 0; t < 4; ++t) {
        const int col = h * DK + t * 16 + nl;
#pragma unroll
        for (int v = 0; v < 8; ++v) {
            const int s = q0 + v + hi * 8;
            AO[(b * SEQ + s) * HID + col] = f2bf(O[t][v] / lrun[v]);
        }
    }
}

// ---------------------------------------------------------------------------
// Kernel 3: output projection  out = AO @ Wo^T + bo (f32), 32x64 per wave.
// ---------------------------------------------------------------------------
__global__ __launch_bounds__(128) void out_gemm_kernel(
    const bf16_t* __restrict__ AO, const bf16_t* __restrict__ Wo,
    const float* __restrict__ bias, float* __restrict__ out) {
    const int lane = threadIdx.x & 31;
    const int widx = threadIdx.x >> 5;
    const int wg   = blockIdx.x * 4 + widx;          // 2048 waves
    const int nt   = wg >> 7;
    const int mt   = wg & 127;
    const int m0   = mt * 32;
    const int n0   = nt * 64;

    v8f acc[2][4];
#pragma unroll
    for (int r = 0; r < 2; ++r)
#pragma unroll
        for (int t = 0; t < 4; ++t) acc[r][t] = (v8f){0,0,0,0,0,0,0,0};

    for (int k0 = 0; k0 < HID; k0 += 32) {
        const v16bf a0 = load_frag(AO, HID, m0,      k0, lane);
        const v16bf a1 = load_frag(AO, HID, m0 + 16, k0, lane);
#pragma unroll
        for (int t = 0; t < 4; ++t) {
            if (k0 + 32 < HID)
                __builtin_prefetch(Wo + (n0 + t * 16 + (lane & 15)) * HID + k0 + 32, 0, 3);
            const v16bf b = load_frag(Wo, HID, n0 + t * 16, k0, lane);
            acc[0][t] = wmma_bf16(a0, b, acc[0][t]);
            acc[1][t] = wmma_bf16(a1, b, acc[1][t]);
        }
    }

    const int hi = lane >> 4;
    const int nl = lane & 15;
#pragma unroll
    for (int t = 0; t < 4; ++t) {
        const int n = n0 + t * 16 + nl;
        const float bv = bias[n];
#pragma unroll
        for (int r = 0; r < 2; ++r)
#pragma unroll
            for (int v = 0; v < 8; ++v) {
                const int m = m0 + r * 16 + v + hi * 8;
                out[m * HID + n] = acc[r][t][v] + bv;
            }
    }
}

// ---------------------------------------------------------------------------
// Host launcher.  d_in: x, Wq, bq, Wv, bv, Wo, bo  (all f32).
// Workspace: x_bf16(8M) | Wq(2M) | Wv(2M) | Wo(2M) | Q[b,h,s,d](8M)
//            | Vt[b,h,d,s](8M) | AO[b,s,hid](8M)
// ---------------------------------------------------------------------------
extern "C" void kernel_launch(void* const* d_in, const int* in_sizes, int n_in,
                              void* d_out, int out_size, void* d_ws, size_t ws_size,
                              hipStream_t stream) {
    const float* x  = (const float*)d_in[0];
    const float* Wq = (const float*)d_in[1];
    const float* bq = (const float*)d_in[2];
    const float* Wv = (const float*)d_in[3];
    const float* bv = (const float*)d_in[4];
    const float* Wo = (const float*)d_in[5];
    const float* bo = (const float*)d_in[6];

    char* ws = (char*)d_ws;
    const size_t SZ_X = (size_t)MTOT * HID * sizeof(bf16_t);   // 8 MiB
    const size_t SZ_W = (size_t)HID * HID * sizeof(bf16_t);    // 2 MiB
    bf16_t* xbf  = (bf16_t*)(ws);
    bf16_t* wqbf = (bf16_t*)(ws + SZ_X);
    bf16_t* wvbf = (bf16_t*)(ws + SZ_X + SZ_W);
    bf16_t* wobf = (bf16_t*)(ws + SZ_X + 2 * SZ_W);
    bf16_t* Qbf  = (bf16_t*)(ws + SZ_X + 3 * SZ_W);
    bf16_t* Vtbf = (bf16_t*)(ws + 2 * SZ_X + 3 * SZ_W);
    bf16_t* AObf = (bf16_t*)(ws + 3 * SZ_X + 3 * SZ_W);

    f32_to_bf16_kernel<<<512, 256, 0, stream>>>(x,  xbf,  (MTOT * HID) / 4);
    f32_to_bf16_kernel<<<256, 256, 0, stream>>>(Wq, wqbf, (HID * HID) / 4);
    f32_to_bf16_kernel<<<256, 256, 0, stream>>>(Wv, wvbf, (HID * HID) / 4);
    f32_to_bf16_kernel<<<256, 256, 0, stream>>>(Wo, wobf, (HID * HID) / 4);

    proj_gemm_kernel<<<512, 128, 0, stream>>>(xbf, wqbf, bq, Qbf, 0);
    proj_gemm_kernel<<<512, 128, 0, stream>>>(xbf, wvbf, bv, Vtbf, 1);

    flash_attn_kernel<<<1024, 128, 0, stream>>>(Qbf, Vtbf, AObf);

    out_gemm_kernel<<<512, 128, 0, stream>>>(AObf, wobf, bo, (float*)d_out);
}
